// Attention_41188736368809
// MI455X (gfx1250) — compile-verified, matrix operand order checked by reading, not run on hardware
//
#include <hip/hip_runtime.h>
#include <hip/hip_bf16.h>

// ---------------------------------------------------------------------------
// Problem constants (from reference): DIM=192, HEADS=6, WIN=8, SHIFT=4
//   B=2, H=W=256 -> wb = 2*32*32 = 2048 windows, N = 64 pixels/window
//   qkv channels = 576, window-attn chans 0..287, channel-attn chans 288..575
// ---------------------------------------------------------------------------
#define CC      192
#define QC      576          // 3*CC
#define HEADS   6
#define DD      16           // head dim
#define NWIN    64           // pixels per window
#define WB      2048         // total windows
#define PERWIN  (QC * NWIN)  // 36864 qkv elements / window
#define OUTWIN  (CC * NWIN)  // 12288 concat elements / window

typedef __attribute__((ext_vector_type(16))) __bf16 v16bf;
typedef __attribute__((ext_vector_type(8)))  __bf16 v8bf;
typedef __attribute__((ext_vector_type(8)))  float  v8f;

// Workspace layout (bytes). dwconv is fused into the qkv GEMM, so only the
// post-conv qkv tensor (bf16) and the attention output (f32) hit HBM.
static constexpr size_t OFF_QKV1 = 0;                          // bf16 [2048*576*64]
static constexpr size_t SZ_QKV   = (size_t)WB * PERWIN * 2;    // 150,994,944
static constexpr size_t OFF_ATTN = OFF_QKV1 + SZ_QKV;          // f32 [2048*192*64]
static constexpr size_t SZ_ATTN  = (size_t)WB * OUTWIN * 4;    // 100,663,296
static constexpr size_t OFF_WQ   = OFF_ATTN + SZ_ATTN;         // bf16 576x192
static constexpr size_t OFF_WP   = OFF_WQ + (size_t)QC * CC * 2;   // bf16 192x192
static constexpr size_t OFF_BIAS = OFF_WP + (size_t)CC * CC * 2;   // f32 225x6

__device__ __forceinline__ v8f zero8() {
    v8f z;
#pragma unroll
    for (int i = 0; i < 8; ++i) z[i] = 0.f;
    return z;
}

// Load one 16x32 bf16 WMMA fragment half for this lane.
// 16-bit A/B VGPR layout: element i (0..15): K = (i/8)*16 + half*8 + (i%8).
// Caller passes p = rowBase + kstep*32 + half*8; group0 at p, group1 at p+16.
__device__ __forceinline__ v16bf ldfrag(const __bf16* p) {
    v8bf lo = *(const v8bf*)(p);
    v8bf hi = *(const v8bf*)(p + 16);
    v16bf r;
#pragma unroll
    for (int i = 0; i < 8; ++i) { r[i] = lo[i]; r[i + 8] = hi[i]; }
    return r;
}

__device__ __forceinline__ v8f wmma_bf16(v16bf a, v16bf b, v8f c) {
    return __builtin_amdgcn_wmma_f32_16x16x32_bf16(false, a, false, b,
                                                   (short)0, c, false, false);
}

// ---------------------------------------------------------------------------
// Kernel 0a: convert qkv_w (576x192) and proj_w (192x192) f32 -> bf16
// ---------------------------------------------------------------------------
__global__ __launch_bounds__(256) void k_convert(const float* qkv_w,
                                                 const float* proj_w,
                                                 __bf16* wq, __bf16* wp) {
    int idx = blockIdx.x * 256 + threadIdx.x;
    if (idx < QC * CC) wq[idx] = (__bf16)qkv_w[idx];
    int j = idx - QC * CC;
    if (j >= 0 && j < CC * CC) wp[j] = (__bf16)proj_w[j];
}

// ---------------------------------------------------------------------------
// Kernel 0b: CPB MLP -> bias_table[225][6]
// coords: e=(i*15+j), t = ((i-7)*8/7, (j-7)*8/7), f = sign*log2(|t|+1)/log2(8)
// ---------------------------------------------------------------------------
__global__ __launch_bounds__(256) void k_cpb(const float* w1, const float* b1,
                                             const float* w2, float* biasT) {
    int e = threadIdx.x;
    if (e >= 225) return;
    float r0 = (float)(e / 15 - 7) * (8.0f / 7.0f);
    float r1 = (float)(e % 15 - 7) * (8.0f / 7.0f);
    float f0 = copysignf(log2f(fabsf(r0) + 1.f) * (1.f / 3.f), r0);
    float f1 = copysignf(log2f(fabsf(r1) + 1.f) * (1.f / 3.f), r1);
    float acc[HEADS];
#pragma unroll
    for (int h = 0; h < HEADS; ++h) acc[h] = 0.f;
    for (int k = 0; k < 512; ++k) {
        float hk = fmaf(w1[2 * k], f0, fmaf(w1[2 * k + 1], f1, b1[k]));
        hk = fmaxf(hk, 0.f);
#pragma unroll
        for (int h = 0; h < HEADS; ++h) acc[h] = fmaf(w2[h * 512 + k], hk, acc[h]);
    }
#pragma unroll
    for (int h = 0; h < HEADS; ++h) biasT[e * HEADS + h] = acc[h];
}

// ---------------------------------------------------------------------------
// Kernel 1: shifted-window gather + qkv GEMM (576x192)x(192x64) per window,
// FUSED with the per-window zero-padded 3x3 depthwise conv.
// One 8-wave block per window. X staged to LDS n-major bf16 (B frags = two
// ds_read_b128). GEMM runs in two 288-channel passes, results parked in LDS,
// depthwise conv consumes them from LDS and writes only post-conv qkv to HBM
// (saves a full 151MB write + 151MB read round trip vs a separate kernel).
// LDS: 24KB (Xs) + 36KB (Qs) = 60KB.
// ---------------------------------------------------------------------------
__global__ __launch_bounds__(256) void k_qkv_dw(const float* __restrict__ x,
                                                const __bf16* __restrict__ Wq,
                                                const float* __restrict__ dww,
                                                __bf16* __restrict__ qkv1) {
    __shared__ alignas(16) __bf16 Xs[NWIN * CC];      // [n][c], 24KB
    __shared__ alignas(16) __bf16 Qs[288 * NWIN];     // [o_local][n], 36KB
    const int win = blockIdx.x;
    const int b  = win >> 10;
    const int wi = win & 1023;
    const int wh = wi >> 5, ww = wi & 31;
    const int tid = threadIdx.x;

    for (int e = tid; e < CC * NWIN; e += 256) {
        int n = e & 63, c = e >> 6;
        int i = n >> 3, j = n & 7;
        int y  = (wh * 8 + i + 4) & 255;   // roll(-4) + window
        int xx = (ww * 8 + j + 4) & 255;
        Xs[n * CC + c] = (__bf16)x[(((size_t)b * CC + c) * 256 + y) * 256 + xx];
    }
    __syncthreads();

    const int wave = tid >> 5, lane = tid & 31;
    const int lm = lane & 15, half = lane >> 4;
    __bf16* outp = qkv1 + (size_t)win * PERWIN;

#pragma unroll 1
    for (int pass = 0; pass < 2; ++pass) {
        // ---- GEMM for channels [pass*288, pass*288+288) -> Qs ----
        for (int mt = wave; mt < 18; mt += 8) {
            const __bf16* arow = Wq + (size_t)((pass * 18 + mt) * 16 + lm) * CC;
            if (mt + 8 < 18)   // hint next weight tile row into cache
                __builtin_prefetch(Wq + (size_t)((pass * 18 + mt + 8) * 16 + lm) * CC, 0, 1);
#pragma unroll
            for (int nt = 0; nt < 4; ++nt) {
                const __bf16* bcol = Xs + (nt * 16 + lm) * CC;
                v8f acc = zero8();
#pragma unroll
                for (int ks = 0; ks < 6; ++ks) {
                    int kb = ks * 32 + half * 8;
                    acc = wmma_bf16(ldfrag(arow + kb), ldfrag(bcol + kb), acc);
                }
                int n = nt * 16 + lm;
#pragma unroll
                for (int r = 0; r < 8; ++r) {
                    int ol = mt * 16 + half * 8 + r;   // C/D layout: M = r + 8*half
                    Qs[ol * NWIN + n] = (__bf16)acc[r];
                }
            }
        }
        __syncthreads();

        // ---- depthwise 3x3 (zero pad inside the 8x8 window) from LDS ----
        {
            int n = tid & 63;
            int i = n >> 3, j = n & 7;
            for (int ol = tid >> 6; ol < 288; ol += 4) {
                int o = pass * 288 + ol;
                const float* wk = dww + o * 9;
                const __bf16* q = Qs + ol * NWIN;
                float s = 0.f;
#pragma unroll
                for (int ky = 0; ky < 3; ++ky) {
                    int yy = i + ky - 1;
                    if (yy < 0 || yy > 7) continue;
#pragma unroll
                    for (int kx = 0; kx < 3; ++kx) {
                        int xx = j + kx - 1;
                        if (xx < 0 || xx > 7) continue;
                        s = fmaf(wk[ky * 3 + kx], (float)q[yy * 8 + xx], s);
                    }
                }
                outp[o * NWIN + n] = (__bf16)s;
            }
        }
        __syncthreads();   // Qs reused by next pass
    }
}

// ---------------------------------------------------------------------------
// Kernel 3: window attention. One 2-wave block per (window, head).
//   QK^T : 16 WMMA tiles, K=16 zero-padded to 32
//   fused logit-scale, 16*sigmoid(cpb bias), shift mask; softmax_one
//   attn*V: 8 WMMA tiles, K=64 (2 ksteps)
// ---------------------------------------------------------------------------
__global__ __launch_bounds__(64) void k_winattn(const __bf16* __restrict__ qkv1,
                                                const float* __restrict__ logit_scale,
                                                const float* __restrict__ biasT,
                                                float* __restrict__ attn_out) {
    __shared__ alignas(16) __bf16 qA[NWIN * 32];      // [n][k<=32], 4KB
    __shared__ alignas(16) __bf16 kB[NWIN * 32];      // 4KB
    __shared__ alignas(16) float  attnS[NWIN * NWIN]; // 16KB
    __shared__ alignas(16) __bf16 attnA[NWIN * NWIN]; // 8KB
    __shared__ alignas(16) __bf16 vB[DD * NWIN];      // [d][m], 2KB

    const int job = blockIdx.x;
    const int win = job / HEADS, h = job % HEADS;
    const int wi = win & 1023;
    const int wh = wi >> 5, ww = wi & 31;
    const int tid = threadIdx.x;
    const __bf16* base = qkv1 + (size_t)win * PERWIN;

    // l2norm(q), l2norm(k) over d; transpose into fragment-friendly LDS
    {
        int n = tid;  // 0..63
        float qv[DD], kv[DD], sq = 0.f, sk = 0.f;
#pragma unroll
        for (int d = 0; d < DD; ++d) {
            qv[d] = (float)base[(h * DD + d) * NWIN + n];
            kv[d] = (float)base[(96 + h * DD + d) * NWIN + n];
            sq = fmaf(qv[d], qv[d], sq);
            sk = fmaf(kv[d], kv[d], sk);
        }
        float iq = 1.f / fmaxf(sqrtf(sq), 1e-12f);
        float ik = 1.f / fmaxf(sqrtf(sk), 1e-12f);
#pragma unroll
        for (int d = 0; d < DD; ++d) {
            qA[n * 32 + d] = (__bf16)(qv[d] * iq);
            qA[n * 32 + 16 + d] = (__bf16)0.f;
            kB[n * 32 + d] = (__bf16)(kv[d] * ik);
            kB[n * 32 + 16 + d] = (__bf16)0.f;
            vB[d * NWIN + n] = base[(192 + h * DD + d) * NWIN + n];
        }
    }
    __syncthreads();

    const float scale = __expf(fminf(logit_scale[h], 4.6051702f)); // log(100)
    const int wave = tid >> 5, lane = tid & 31;
    const int lm = lane & 15, half = lane >> 4;

    // ---- attn = scale * qn kn^T + 16*sigmoid(rpb) + mask ----
#pragma unroll
    for (int mi = 0; mi < 2; ++mi) {
        int mt = wave * 2 + mi;
        v16bf a = ldfrag(qA + (mt * 16 + lm) * 32 + half * 8);
#pragma unroll
        for (int nt = 0; nt < 4; ++nt) {
            v16bf bb = ldfrag(kB + (nt * 16 + lm) * 32 + half * 8);
            v8f acc = wmma_bf16(a, bb, zero8());
            int m = nt * 16 + lm;
            int i2 = m >> 3, j2 = m & 7;
            int rh2 = (wh * 8 + i2 < 248) ? 0 : ((wh * 8 + i2 < 252) ? 1 : 2);
            int rw2 = (ww * 8 + j2 < 248) ? 0 : ((ww * 8 + j2 < 252) ? 1 : 2);
            int cm = rh2 * 3 + rw2;
#pragma unroll
            for (int r = 0; r < 8; ++r) {
                int n = mt * 16 + half * 8 + r;
                int i1 = n >> 3, j1 = n & 7;
                int rh1 = (wh * 8 + i1 < 248) ? 0 : ((wh * 8 + i1 < 252) ? 1 : 2);
                int rw1 = (ww * 8 + j1 < 248) ? 0 : ((ww * 8 + j1 < 252) ? 1 : 2);
                float mask = ((rh1 * 3 + rw1) == cm) ? 0.f : -100.f;
                int idx = (i1 - i2 + 7) * 15 + (j1 - j2 + 7);
                float bias = biasT[idx * HEADS + h];
                float sig = 16.f / (1.f + __expf(-bias));
                attnS[n * NWIN + m] = fmaf(acc[r], scale, sig) + mask;
            }
        }
    }
    __syncthreads();

    // ---- softmax_one per row ----
    {
        int n = tid;
        float mx = -1e30f;
#pragma unroll 8
        for (int m = 0; m < NWIN; ++m) mx = fmaxf(mx, attnS[n * NWIN + m]);
        float s = 0.f;
#pragma unroll 8
        for (int m = 0; m < NWIN; ++m) s += __expf(attnS[n * NWIN + m] - mx);
        float inv = 1.f / (1.f + s);
#pragma unroll 8
        for (int m = 0; m < NWIN; ++m)
            attnA[n * NWIN + m] = (__bf16)(__expf(attnS[n * NWIN + m] - mx) * inv);
    }
    __syncthreads();

    // ---- out = attn (64x64) x V (64x16) ----
    float* outp = attn_out + (size_t)win * OUTWIN;
#pragma unroll
    for (int mi = 0; mi < 2; ++mi) {
        int mt = wave * 2 + mi;
        v8f acc = zero8();
#pragma unroll
        for (int ks = 0; ks < 2; ++ks) {
            int kb = ks * 32 + half * 8;
            v16bf a  = ldfrag(attnA + (mt * 16 + lm) * NWIN + kb);
            v16bf bb = ldfrag(vB + lm * NWIN + kb);
            acc = wmma_bf16(a, bb, acc);
        }
        int d = lm;
#pragma unroll
        for (int r = 0; r < 8; ++r) {
            int n = mt * 16 + half * 8 + r;
            outp[(h * DD + d) * NWIN + n] = acc[r];
        }
    }
}

// ---------------------------------------------------------------------------
// Kernel 4: channel attention. One wave per (window, head).
//   a = l2norm_n(q) l2norm_n(k)^T  (16x16, K=64 -> 2 WMMA)
//   softmax_one over rows; out = a (K=16 padded) x v (16x64 -> 4 WMMA)
// ---------------------------------------------------------------------------
__global__ __launch_bounds__(32) void k_chattn(const __bf16* __restrict__ qkv1,
                                               float* __restrict__ attn_out) {
    __shared__ alignas(16) __bf16 qA[DD * NWIN];   // [c][n], 2KB
    __shared__ alignas(16) __bf16 kB[DD * NWIN];   // 2KB
    __shared__ alignas(16) __bf16 vB[NWIN * 32];   // [n][e<=32], 4KB
    __shared__ alignas(16) float  aS[DD * DD];     // 1KB
    __shared__ alignas(16) __bf16 aA[DD * 32];     // 1KB

    const int job = blockIdx.x;
    const int win = job / HEADS, h = job % HEADS;
    const int lane = threadIdx.x;
    const int lm = lane & 15, half = lane >> 4;
    const __bf16* base = qkv1 + (size_t)win * PERWIN;

    {   // l2norm over n (row length 64); q rows -> lanes 0..15, k -> 16..31
        int d = lm;
        int o = half ? (384 + h * DD + d) : (288 + h * DD + d);
        float v[NWIN], s = 0.f;
#pragma unroll
        for (int n = 0; n < NWIN; ++n) {
            v[n] = (float)base[o * NWIN + n];
            s = fmaf(v[n], v[n], s);
        }
        float inv = 1.f / fmaxf(sqrtf(s), 1e-12f);
        __bf16* dst = half ? (kB + d * NWIN) : (qA + d * NWIN);
#pragma unroll
        for (int n = 0; n < NWIN; ++n) dst[n] = (__bf16)(v[n] * inv);
        // v staged transposed with K(e) zero-padded to 32
#pragma unroll
        for (int t = 0; t < 2; ++t) {
            int n = lane + t * 32;
#pragma unroll
            for (int e = 0; e < DD; ++e) {
                vB[n * 32 + e] = base[(480 + h * DD + e) * NWIN + n];
                vB[n * 32 + 16 + e] = (__bf16)0.f;
            }
        }
    }
    __syncthreads();

    {   // a = qn kn^T, K = 64
        v8f acc = zero8();
#pragma unroll
        for (int ks = 0; ks < 2; ++ks) {
            int kb = ks * 32 + half * 8;
            acc = wmma_bf16(ldfrag(qA + lm * NWIN + kb),
                            ldfrag(kB + lm * NWIN + kb), acc);
        }
#pragma unroll
        for (int r = 0; r < 8; ++r) aS[(half * 8 + r) * DD + lm] = acc[r];
    }
    __syncthreads();

    if (lane < DD) {  // softmax_one per row (16 wide)
        int c = lane;
        float mx = -1e30f;
#pragma unroll
        for (int e = 0; e < DD; ++e) mx = fmaxf(mx, aS[c * DD + e]);
        float s = 0.f;
#pragma unroll
        for (int e = 0; e < DD; ++e) s += __expf(aS[c * DD + e] - mx);
        float inv = 1.f / (1.f + s);
#pragma unroll
        for (int e = 0; e < DD; ++e)
            aA[c * 32 + e] = (__bf16)(__expf(aS[c * DD + e] - mx) * inv);
#pragma unroll
        for (int e = 0; e < DD; ++e) aA[c * 32 + 16 + e] = (__bf16)0.f;
    }
    __syncthreads();

    // out = a (16x16, pad K to 32) x v (16x64)
    float* outp = attn_out + (size_t)win * OUTWIN;
    v16bf a = ldfrag(aA + lm * 32 + half * 8);
#pragma unroll
    for (int nt = 0; nt < 4; ++nt) {
        v16bf bb = ldfrag(vB + (nt * 16 + lm) * 32 + half * 8);
        v8f acc = wmma_bf16(a, bb, zero8());
        int n = nt * 16 + lm;
#pragma unroll
        for (int r = 0; r < 8; ++r) {
            int c = half * 8 + r;
            outp[(96 + h * DD + c) * NWIN + n] = acc[r];
        }
    }
}

// ---------------------------------------------------------------------------
// Kernel 5: proj GEMM (192x192)x(192x64) per window, fused un-window + roll(+4)
// ---------------------------------------------------------------------------
__global__ __launch_bounds__(256) void k_proj(const float* __restrict__ attn_out,
                                              const __bf16* __restrict__ Wp,
                                              float* __restrict__ out) {
    __shared__ alignas(16) __bf16 Us[NWIN * CC];  // [n][c], 24KB
    const int win = blockIdx.x;
    const int b  = win >> 10;
    const int wi = win & 1023;
    const int wh = wi >> 5, ww = wi & 31;
    const int tid = threadIdx.x;
    const float* src = attn_out + (size_t)win * OUTWIN;

    for (int e = tid; e < OUTWIN; e += 256) {
        int n = e & 63, c = e >> 6;
        Us[n * CC + c] = (__bf16)src[c * NWIN + n];
    }
    __syncthreads();

    const int wave = tid >> 5, lane = tid & 31;
    const int lm = lane & 15, half = lane >> 4;

#pragma unroll
    for (int s = 0; s < 6; ++s) {
        int t = wave * 6 + s;          // 48 tiles: 12 mt x 4 nt
        int mt = t >> 2, nt = t & 3;
        const __bf16* arow = Wp + (size_t)(mt * 16 + lm) * CC;
        const __bf16* bcol = Us + (nt * 16 + lm) * CC;
        v8f acc = zero8();
#pragma unroll
        for (int ks = 0; ks < 6; ++ks) {
            int kb = ks * 32 + half * 8;
            acc = wmma_bf16(ldfrag(arow + kb), ldfrag(bcol + kb), acc);
        }
        int n = nt * 16 + lm;
        int i = n >> 3, j = n & 7;
        int y  = (wh * 8 + i + 4) & 255;   // roll(+4,+4) back
        int xx = (ww * 8 + j + 4) & 255;
#pragma unroll
        for (int r = 0; r < 8; ++r) {
            int o = mt * 16 + half * 8 + r;
            out[(((size_t)b * CC + o) * 256 + y) * 256 + xx] = acc[r];
        }
    }
}

// ---------------------------------------------------------------------------
extern "C" void kernel_launch(void* const* d_in, const int* in_sizes, int n_in,
                              void* d_out, int out_size, void* d_ws, size_t ws_size,
                              hipStream_t stream) {
    const float* x           = (const float*)d_in[0];
    const float* qkv_w       = (const float*)d_in[1];
    const float* dw_w        = (const float*)d_in[2];
    const float* proj_w      = (const float*)d_in[3];
    const float* logit_scale = (const float*)d_in[4];
    const float* cpb_w1      = (const float*)d_in[5];
    const float* cpb_b1      = (const float*)d_in[6];
    const float* cpb_w2      = (const float*)d_in[7];

    char* ws = (char*)d_ws;
    __bf16* qkv1   = (__bf16*)(ws + OFF_QKV1);
    float*  attn_o = (float*)(ws + OFF_ATTN);
    __bf16* wq     = (__bf16*)(ws + OFF_WQ);
    __bf16* wp     = (__bf16*)(ws + OFF_WP);
    float*  biasT  = (float*)(ws + OFF_BIAS);
    float*  out    = (float*)d_out;

    k_convert<<<(QC * CC + CC * CC + 255) / 256, 256, 0, stream>>>(qkv_w, proj_w, wq, wp);
    k_cpb<<<1, 256, 0, stream>>>(cpb_w1, cpb_b1, cpb_w2, biasT);
    k_qkv_dw<<<WB, 256, 0, stream>>>(x, wq, dw_w, qkv1);
    k_winattn<<<WB * HEADS, 64, 0, stream>>>(qkv1, logit_scale, biasT, attn_o);
    k_chattn<<<WB * HEADS, 32, 0, stream>>>(qkv1, attn_o);
    k_proj<<<WB, 256, 0, stream>>>(attn_o, wp, out);
}
